// MoELayer_10952166604905
// MI455X (gfx1250) — matrix operand order repaired and checked by reading, not compiled, not run on hardware
//
#include <hip/hip_runtime.h>

// ---------------------------------------------------------------------------
// MoE block-sparse layer for gfx1250 (MI455X), bf16x3-split WMMA GEMM with
// double-buffered GLOBAL_LOAD_ASYNC_TO_LDS staging.
//   out[b,c] = sum_e gate[b,e] * blockmask[b/1024,e] * sum_k x[b,k]*W[k, e*1024+c]
// ---------------------------------------------------------------------------

#define D_MODEL 1024
#define NUM_EXPERTS 8
#define BATCH 8192

typedef unsigned int  u32x4   __attribute__((ext_vector_type(4)));
typedef unsigned int  u32x8   __attribute__((ext_vector_type(8)));
typedef float         f32x8   __attribute__((ext_vector_type(8)));
typedef __bf16        bf16x16 __attribute__((ext_vector_type(16)));

__device__ __forceinline__ unsigned short f32_to_bf16(float f) {
  unsigned int u = __float_as_uint(f);
  u += 0x7FFFu + ((u >> 16) & 1u);          // round-to-nearest-even
  return (unsigned short)(u >> 16);
}
__device__ __forceinline__ float bf16_to_f32(unsigned short h) {
  return __uint_as_float(((unsigned int)h) << 16);
}

// ---- CDNA5 async memory->LDS copy (ASYNCcnt-tracked, no VGPR data path) ---
__device__ __forceinline__ void async_copy_b128(const unsigned short* gsrc,
                                                unsigned short* ldst) {
  unsigned lofs = (unsigned)(unsigned long long)ldst;  // flat->LDS: addr[31:0]
  asm volatile("global_load_async_to_lds_b128 %0, %1, off"
               :: "v"(lofs), "v"(gsrc)
               : "memory");
}
__device__ __forceinline__ void wait_async0() {
  asm volatile("s_wait_asynccnt 0" ::: "memory");
}

// ---- split x (row-major, unchanged layout) --------------------------------
__global__ __launch_bounds__(256)
void conv_split_x(const float* __restrict__ src,
                  unsigned short* __restrict__ hi,
                  unsigned short* __restrict__ lo) {
  size_t i = (size_t)blockIdx.x * 256 + threadIdx.x;
  float v = src[i];
  unsigned short h = f32_to_bf16(v);
  float rem = v - bf16_to_f32(h);
  hi[i] = h;
  lo[i] = f32_to_bf16(rem);
}

// ---- split + transpose W: W[k][e*1024+n] -> Wt[e*1024+n][k] ---------------
__global__ __launch_bounds__(256)
void conv_split_wT(const float* __restrict__ W,
                   unsigned short* __restrict__ hi,
                   unsigned short* __restrict__ lo) {
  size_t i = (size_t)blockIdx.x * 256 + threadIdx.x;   // i = k*8192 + j
  int k = (int)(i >> 13);
  int j = (int)(i & 8191);
  float v = W[i];
  unsigned short h = f32_to_bf16(v);
  float rem = v - bf16_to_f32(h);
  size_t o = (size_t)j * D_MODEL + k;
  hi[o] = h;
  lo[o] = f32_to_bf16(rem);
}

// ---- gating: one wave32 per row; softmax over 8, top-2 block flags --------
__global__ __launch_bounds__(256)
void moe_gate(const float* __restrict__ x,
              const float* __restrict__ gw,
              const float* __restrict__ gb,
              float* __restrict__ gat,
              int* __restrict__ flags) {
  const int lane = threadIdx.x & 31;
  const int wave = threadIdx.x >> 5;
  const int b = blockIdx.x * 8 + wave;

  float acc[NUM_EXPERTS];
#pragma unroll
  for (int e = 0; e < NUM_EXPERTS; ++e) acc[e] = 0.f;

  const float* xr = x + (size_t)b * D_MODEL;
  for (int i = lane; i < D_MODEL; i += 32) {
    float xv = xr[i];
    const float* g = gw + (size_t)i * NUM_EXPERTS;
#pragma unroll
    for (int e = 0; e < NUM_EXPERTS; ++e) acc[e] += xv * g[e];
  }
#pragma unroll
  for (int off = 16; off >= 1; off >>= 1) {
#pragma unroll
    for (int e = 0; e < NUM_EXPERTS; ++e) acc[e] += __shfl_xor(acc[e], off, 32);
  }
  float lg[NUM_EXPERTS], mx = -1e30f;
#pragma unroll
  for (int e = 0; e < NUM_EXPERTS; ++e) { lg[e] = acc[e] + gb[e]; mx = fmaxf(mx, lg[e]); }
  float p[NUM_EXPERTS], s = 0.f;
#pragma unroll
  for (int e = 0; e < NUM_EXPERTS; ++e) { p[e] = __expf(lg[e] - mx); s += p[e]; }
  float inv = 1.f / s;
  if (lane == 0) {
#pragma unroll
    for (int e = 0; e < NUM_EXPERTS; ++e) gat[(size_t)b * NUM_EXPERTS + e] = p[e] * inv;
    int e1 = 0;
#pragma unroll
    for (int e = 1; e < NUM_EXPERTS; ++e) if (p[e] > p[e1]) e1 = e;
    int e2 = (e1 == 0) ? 1 : 0;
#pragma unroll
    for (int e = 0; e < NUM_EXPERTS; ++e) if (e != e1 && p[e] > p[e2]) e2 = e;
    int rb = b >> 10;
    atomicOr(&flags[rb * NUM_EXPERTS + e1], 1);
    atomicOr(&flags[rb * NUM_EXPERTS + e2], 1);
  }
}

// ---- main GEMM + expert combine: 128x128 tile / WG, 8 waves ---------------
// wave (wm,wn) owns 64x32; 4x2 subtiles of 16x16; 3 wmmas per subtile per
// 32-wide k-tile (bf16x3: hh + lh + hl). Double-buffered async LDS staging.
union AOp { u32x4 q[2]; bf16x16 v; };
union BOp { u32x8 q;    bf16x16 v; };

__global__ __launch_bounds__(256)
void moe_gemm(const unsigned short* __restrict__ xh,
              const unsigned short* __restrict__ xl,
              const unsigned short* __restrict__ wth,
              const unsigned short* __restrict__ wtl,
              const float* __restrict__ gat,
              const int* __restrict__ flags,
              float* __restrict__ out) {
  __shared__ __align__(64) unsigned short s_xh[2][128 * 32];
  __shared__ __align__(64) unsigned short s_xl[2][128 * 32];
  __shared__ __align__(64) unsigned short s_wh[2][128 * 32];
  __shared__ __align__(64) unsigned short s_wl[2][128 * 32];

  const int tid  = threadIdx.x;
  const int lane = tid & 31;
  const int wave = tid >> 5;
  const int wm = wave & 1;       // 0..1 -> 64-row half
  const int wn = wave >> 1;      // 0..3 -> 32-col quarter
  const int halfK = lane >> 4;   // lane half selects K offset (contiguous k = 16h + s)
  const int mlane = lane & 15;

  const int bm = blockIdx.x * 128;
  const int bn = blockIdx.y * 128;
  const int rb = blockIdx.x >> 3;   // 1024-row block index

  const f32x8 zv = {0.f, 0.f, 0.f, 0.f, 0.f, 0.f, 0.f, 0.f};
  f32x8 accO[4][2];
#pragma unroll
  for (int mi = 0; mi < 4; ++mi)
#pragma unroll
    for (int ni = 0; ni < 2; ++ni) accO[mi][ni] = zv;

  for (int e = 0; e < NUM_EXPERTS; ++e) {
    if (flags[rb * NUM_EXPERTS + e] == 0) continue;   // block-uniform skip

    // issue async copies of one 32-wide k-tile (4 bf16 tiles) into buffer buf
    auto stage = [&](int buf, int k0) {
#pragma unroll
      for (int it = 0; it < 2; ++it) {
        int c   = tid + it * 256;      // 512 16B-chunks per tile
        int row = c >> 2;
        int cc  = c & 3;
        int lofs = row * 32 + cc * 8;
        size_t gx = (size_t)(bm + row) * D_MODEL + k0 + cc * 8;
        size_t gw = (size_t)(e * D_MODEL + bn + row) * D_MODEL + k0 + cc * 8;
        async_copy_b128(&xh[gx],  &s_xh[buf][lofs]);
        async_copy_b128(&xl[gx],  &s_xl[buf][lofs]);
        async_copy_b128(&wth[gw], &s_wh[buf][lofs]);
        async_copy_b128(&wtl[gw], &s_wl[buf][lofs]);
      }
    };

    f32x8 acc[4][2];
#pragma unroll
    for (int mi = 0; mi < 4; ++mi)
#pragma unroll
      for (int ni = 0; ni < 2; ++ni) acc[mi][ni] = zv;

    stage(0, 0);
    wait_async0();
    __syncthreads();

    for (int kt = 0; kt < D_MODEL / 32; ++kt) {
      const int cur = kt & 1;
      if (kt + 1 < D_MODEL / 32) stage(cur ^ 1, (kt + 1) * 32);

      // ---- B operands: per-lane column N, 16 contiguous K (32B) ----
      BOp bh[2], bl[2];
#pragma unroll
      for (int ni = 0; ni < 2; ++ni) {
        int nl = wn * 32 + ni * 16 + mlane;
        int bofs = nl * 32 + halfK * 16;
        bh[ni].q = *(const u32x8*)&s_wh[cur][bofs];
        bl[ni].q = *(const u32x8*)&s_wl[cur][bofs];
      }
      // ---- A operands + wmma ----
#pragma unroll
      for (int mi = 0; mi < 4; ++mi) {
        int ml = wm * 64 + mi * 16 + mlane;
        int aofs = ml * 32 + halfK * 16;
        AOp ah, al;
        ah.q[0] = *(const u32x4*)&s_xh[cur][aofs];
        ah.q[1] = *(const u32x4*)&s_xh[cur][aofs + 8];
        al.q[0] = *(const u32x4*)&s_xl[cur][aofs];
        al.q[1] = *(const u32x4*)&s_xl[cur][aofs + 8];
#pragma unroll
        for (int ni = 0; ni < 2; ++ni) {
          acc[mi][ni] = __builtin_amdgcn_wmma_f32_16x16x32_bf16(
              false, ah.v, false, bh[ni].v, (short)0, acc[mi][ni], false, false);
          acc[mi][ni] = __builtin_amdgcn_wmma_f32_16x16x32_bf16(
              false, al.v, false, bh[ni].v, (short)0, acc[mi][ni], false, false);
          acc[mi][ni] = __builtin_amdgcn_wmma_f32_16x16x32_bf16(
              false, ah.v, false, bl[ni].v, (short)0, acc[mi][ni], false, false);
        }
      }

      wait_async0();     // next buffer's async copies have landed in LDS
      __syncthreads();   // all waves done reading cur before it is re-staged
    }

    // ---- gating-weighted combine; C layout: VGPR r -> M=r(+8 for hi lanes) ----
#pragma unroll
    for (int mi = 0; mi < 4; ++mi) {
      int rowBase = bm + wm * 64 + mi * 16 + halfK * 8;
#pragma unroll
      for (int r = 0; r < 8; ++r) {
        float wgt = gat[(size_t)(rowBase + r) * NUM_EXPERTS + e];
        accO[mi][0][r] += wgt * acc[mi][0][r];
        accO[mi][1][r] += wgt * acc[mi][1][r];
      }
    }
  }

  // ---- store fp32 output tile ----
#pragma unroll
  for (int mi = 0; mi < 4; ++mi) {
#pragma unroll
    for (int ni = 0; ni < 2; ++ni) {
#pragma unroll
      for (int r = 0; r < 8; ++r) {
        int row = bm + wm * 64 + mi * 16 + halfK * 8 + r;
        int col = bn + wn * 32 + ni * 16 + mlane;
        out[(size_t)row * D_MODEL + col] = accO[mi][ni][r];
      }
    }
  }
}

// ---------------------------------------------------------------------------
extern "C" void kernel_launch(void* const* d_in, const int* in_sizes, int n_in,
                              void* d_out, int out_size, void* d_ws, size_t ws_size,
                              hipStream_t stream) {
  const float* x  = (const float*)d_in[0];
  const float* W  = (const float*)d_in[1];
  const float* gw = (const float*)d_in[2];
  const float* gb = (const float*)d_in[3];
  float* out = (float*)d_out;

  char* ws = (char*)d_ws;
  const size_t NX = (size_t)BATCH * D_MODEL;        // 8M elements (x and W alike)
  const size_t GAT_OFF = 0;                         // [8192][8] f32 = 256KB
  const size_t FLG_OFF = 256 * 1024;                // [8][8] int
  const size_t XH_OFF  = 256 * 1024 + 4096;
  const size_t XL_OFF  = XH_OFF + NX * 2;
  const size_t WH_OFF  = XL_OFF + NX * 2;
  const size_t WL_OFF  = WH_OFF + NX * 2;           // total ~64.26 MB of ws

  float* gat = (float*)(ws + GAT_OFF);
  int* flags = (int*)(ws + FLG_OFF);
  unsigned short* xh  = (unsigned short*)(ws + XH_OFF);
  unsigned short* xl  = (unsigned short*)(ws + XL_OFF);
  unsigned short* wth = (unsigned short*)(ws + WH_OFF);
  unsigned short* wtl = (unsigned short*)(ws + WL_OFF);

  hipMemsetAsync(flags, 0, NUM_EXPERTS * (BATCH / D_MODEL) * sizeof(int), stream);

  conv_split_x <<<dim3((unsigned)(NX / 256)), dim3(256), 0, stream>>>(x, xh, xl);
  conv_split_wT<<<dim3((unsigned)(NX / 256)), dim3(256), 0, stream>>>(W, wth, wtl);
  moe_gate<<<dim3(BATCH / 8), dim3(256), 0, stream>>>(x, gw, gb, gat, flags);
  moe_gemm<<<dim3(BATCH / 128, D_MODEL / 128), dim3(256), 0, stream>>>(
      xh, xl, wth, wtl, gat, flags, out);
}
